// StateRN_62560493634197
// MI455X (gfx1250) — compile-verified, matrix operand order checked by reading, not run on hardware
//
#include <hip/hip_runtime.h>
#include <hip/hip_bf16.h>
#include <math.h>

typedef __attribute__((ext_vector_type(16))) _Float16 v16h;
typedef __attribute__((ext_vector_type(8)))  _Float16 v8h;
typedef __attribute__((ext_vector_type(8)))  float    v8f;
typedef __attribute__((ext_vector_type(4)))  unsigned int v4u;
typedef __attribute__((ext_vector_type(8)))  int      v8i;
typedef __attribute__((ext_vector_type(4)))  int      v4i;

#define HID 256
#define ROWS_PER_WG 128   // 8 M-tiles; wave grid = 2 (M) x 4 (N)
#define SLABS 5           // 5*128 = 640 padded rows per batch (625 valid)
#define HSTRIDE 264       // halfwords; 528B = 33*16 -> 16B aligned, conflict-free
#define WSTRIDE 40        // halfwords; 80B = 5*16 -> 16B aligned, conflict-free

// ---------------------------------------------------------------------------
// Weight convert: w[k][256] f32 -> wt[n][Kpad] f16 (transposed, zero padded K)
// ---------------------------------------------------------------------------
__global__ void wconv_kernel(const float* __restrict__ w, _Float16* __restrict__ wt,
                             int K, int Kpad) {
    int idx = blockIdx.x * blockDim.x + threadIdx.x;
    int n = idx / Kpad;
    int k = idx - n * Kpad;
    if (n < 256)
        wt[n * Kpad + k] = (k < K) ? (_Float16)w[k * 256 + n] : (_Float16)0.0f;
}

__global__ void zero_kernel(float* __restrict__ p, int n) {
    int i = blockIdx.x * blockDim.x + threadIdx.x;
    if (i < n) p[i] = 0.0f;
}

__device__ __forceinline__ v16h cat16(v8h lo, v8h hi) {
    return __builtin_shufflevector(lo, hi, 0, 1, 2, 3, 4, 5, 6, 7,
                                   8, 9, 10, 11, 12, 13, 14, 15);
}

// Issue a TDM load of one 32-wide K-chunk of transposed weights into LDS.
// Tile: 256 rows x 32 f16 (64B) per row; global row stride = Kd elements.
// LDS padding: 4 DWORDs after every 16 DWORDs -> row stride 40 halfwords.
__device__ __forceinline__ void tdm_stage_chunk(const _Float16* wt, int Kd, int kc,
                                                unsigned int lds_off) {
    unsigned long long ga = (unsigned long long)(wt + kc * 32);
    v4u g0;
    g0[0] = 1u;                                   // count=1, is_restore=0
    g0[1] = lds_off;                              // lds_addr
    g0[2] = (unsigned int)(ga & 0xFFFFFFFFull);   // global_addr[31:0]
    g0[3] = (unsigned int)((ga >> 32) & 0x1FFFFFFull) | (2u << 30); // addr[56:32]|type=2
    v8i g1;
    g1[0] = (int)((1u << 16)      // data_size = 2 bytes
                | (1u << 20)      // pad_enable
                | (3u << 22)      // pad_interval: 16 DWORDs
                | (3u << 25));    // pad_amount:   4 DWORDs
    g1[1] = (int)(((unsigned)Kd & 0xFFFFu) << 16);          // tensor_dim0 lo16
    g1[2] = (int)((((unsigned)Kd >> 16) & 0xFFFFu)
                | (256u << 16));                            // dim0 hi | tensor_dim1 lo
    g1[3] = (int)(0u | (32u << 16));                        // dim1 hi | tile_dim0=32
    g1[4] = (int)(256u);                                    // tile_dim1=256, tile_dim2=0
    g1[5] = (int)((unsigned)Kd);                            // tensor_dim0_stride lo32
    g1[6] = 0;                                              // stride0 hi | stride1 lo
    g1[7] = 0;                                              // stride1 hi
    v4i z4 = {};
    v8i z8 = {};
    __builtin_amdgcn_tensor_load_to_lds(g0, g1, z4, z4, z8, 0);
}

// ---------------------------------------------------------------------------
// g-MLP over all pairs + masked sum into xg[B][256]
// grid.x = B * SLABS, block = 256 (8 waves)
// ---------------------------------------------------------------------------
__global__ __launch_bounds__(256)
void rn_g_kernel(const float* __restrict__ img, const float* __restrict__ qst,
                 const _Float16* __restrict__ wt1, const float* __restrict__ gb1,
                 const _Float16* __restrict__ wt2, const float* __restrict__ gb2,
                 const _Float16* __restrict__ wt3, const float* __restrict__ gb3,
                 const _Float16* __restrict__ wt4, const float* __restrict__ gb4,
                 float* __restrict__ xg)
{
    __shared__ _Float16 hbuf[2][ROWS_PER_WG][HSTRIDE]; // activation ping-pong
    __shared__ _Float16 wch[256][WSTRIDE];             // one 32-wide K-chunk (transposed)
    __shared__ float    xg_lds[256];

    const int tid  = threadIdx.x;
    const int wave = tid >> 5;
    const int lane = tid & 31;
    const int ln   = lane & 15;
    const int hh   = lane >> 4;
    const int wm   = wave >> 2;       // 0..1 : which 4 M-tiles (64 rows)
    const int wn   = wave & 3;        // 0..3 : which 4 N-tiles (64 cols)
    const int blk  = blockIdx.x;
    const int b    = blk / SLABS;
    const int slab = blk - b * SLABS;
    const int mrow = wm * 64;         // first row of this wave's M range
    const int ncol = wn * 64;         // first col of this wave's N range

    xg_lds[tid] = 0.0f;

    // ---- build layer-1 input [128][32]: [obj_c(4) | obj_a(4) | qst(11) | 0-pad] ----
    {
        int row = tid >> 1;
        int cb  = (tid & 1) * 16;
        int p   = slab * ROWS_PER_WG + row;
        int pc  = (p < 625) ? p : 0;
        int a   = pc / 25;
        int c   = pc - a * 25;
        const float* oi = img + (b * 25 + c) * 4;
        const float* oj = img + (b * 25 + a) * 4;
        const float* q  = qst + b * 11;
#pragma unroll
        for (int kk = 0; kk < 16; ++kk) {
            int col = cb + kk;
            float v;
            if (col < 4)       v = oi[col];
            else if (col < 8)  v = oj[col - 4];
            else if (col < 19) v = q[col - 8];
            else               v = 0.0f;
            hbuf[0][row][col] = (_Float16)v;
        }
    }

    const unsigned int wch_off = (unsigned int)(unsigned long long)&wch[0][0];
    const _Float16* wts[4] = {wt1, wt2, wt3, wt4};
    const float*    gbs[4] = {gb1, gb2, gb3, gb4};

#pragma unroll
    for (int l = 0; l < 4; ++l) {
        _Float16 (*hin)[HSTRIDE]  = hbuf[l & 1];
        _Float16 (*hout)[HSTRIDE] = hbuf[(l & 1) ^ 1];
        const _Float16* wt = wts[l];
        const float*    gb = gbs[l];
        const int Kd  = (l == 0) ? 32 : HID;
        const int nkc = Kd >> 5;

        v8f acc[4][4] = {};

        for (int kc = 0; kc < nkc; ++kc) {
            __syncthreads();   // previous chunk fully consumed by all waves
            if (wave == 0) {   // TDM: DMA 16KB weight chunk into LDS (async DMA engine)
                tdm_stage_chunk(wt, Kd, kc, wch_off);
                __builtin_amdgcn_s_wait_tensorcnt(0);
            }
            __syncthreads();

            // A fragments: 4 M-tiles, lane = M row, two contiguous b128 loads each
            v16h af[4];
#pragma unroll
            for (int mt = 0; mt < 4; ++mt) {
                const int m = mrow + mt * 16 + ln;
                v8h a0 = *(const v8h*)&hin[m][kc * 32 + 8 * hh];
                v8h a1 = *(const v8h*)&hin[m][kc * 32 + 16 + 8 * hh];
                af[mt] = cat16(a0, a1);
            }
            // B fragments: 4 N-tiles, lane = N column (weights pre-transposed)
            v16h bf[4];
#pragma unroll
            for (int nt = 0; nt < 4; ++nt) {
                const int n = ncol + nt * 16 + ln;
                v8h b0 = *(const v8h*)&wch[n][8 * hh];
                v8h b1 = *(const v8h*)&wch[n][16 + 8 * hh];
                bf[nt] = cat16(b0, b1);
            }
#pragma unroll
            for (int mt = 0; mt < 4; ++mt)
#pragma unroll
                for (int nt = 0; nt < 4; ++nt)
                    acc[mt][nt] = __builtin_amdgcn_wmma_f32_16x16x32_f16(
                        false, af[mt], false, bf[nt], (short)0, acc[mt][nt],
                        false, false);
        }

        if (l < 3) {
            // bias + ReLU -> f16 into the other activation buffer
#pragma unroll
            for (int mt = 0; mt < 4; ++mt)
#pragma unroll
                for (int nt = 0; nt < 4; ++nt) {
                    const int n = ncol + nt * 16 + ln;
                    const float bias = gb[n];
#pragma unroll
                    for (int r = 0; r < 8; ++r) {
                        float v = acc[mt][nt][r] + bias;
                        v = v > 0.0f ? v : 0.0f;
                        hout[mrow + mt * 16 + r + 8 * hh][n] = (_Float16)v;
                    }
                }
        } else {
            // final layer: bias + ReLU + masked row-sum into shared accumulator
#pragma unroll
            for (int mt = 0; mt < 4; ++mt) {
                const int prow = slab * ROWS_PER_WG + mrow + mt * 16;
#pragma unroll
                for (int nt = 0; nt < 4; ++nt) {
                    const int n = ncol + nt * 16 + ln;
                    const float bias = gb[n];
                    float s = 0.0f;
#pragma unroll
                    for (int r = 0; r < 8; ++r) {
                        float v = acc[mt][nt][r] + bias;
                        v = v > 0.0f ? v : 0.0f;
                        if (prow + r + 8 * hh < 625) s += v;
                    }
                    atomicAdd(&xg_lds[n], s);   // ds_add_f32
                }
            }
        }
    }

    __syncthreads();
    atomicAdd(&xg[b * 256 + tid], xg_lds[tid]);  // global_atomic_add_f32
}

// ---------------------------------------------------------------------------
// f-MLP + log_softmax: 4 batch rows per block, 256 threads
// ---------------------------------------------------------------------------
__global__ __launch_bounds__(256)
void rn_f_kernel(const float* __restrict__ xg,
                 const float* __restrict__ fw1, const float* __restrict__ fb1,
                 const float* __restrict__ fw2, const float* __restrict__ fb2,
                 const float* __restrict__ fw3, const float* __restrict__ fb3,
                 const float* __restrict__ fw4, const float* __restrict__ fb4,
                 float* __restrict__ out)
{
    __shared__ float xs[4][256];
    __shared__ float lg[4][10];
    const int tid = threadIdx.x;
    const int b0  = blockIdx.x * 4;

#pragma unroll
    for (int r = 0; r < 4; ++r) xs[r][tid] = xg[(b0 + r) * 256 + tid];
    __syncthreads();

    const float* ws[3] = {fw1, fw2, fw3};
    const float* bs[3] = {fb1, fb2, fb3};
    for (int l = 0; l < 3; ++l) {
        const float* w = ws[l];
        float a0 = bs[l][tid], a1 = a0, a2 = a0, a3 = a0;
        for (int k = 0; k < 256; ++k) {
            float wv = w[k * 256 + tid];
            a0 = fmaf(xs[0][k], wv, a0);
            a1 = fmaf(xs[1][k], wv, a1);
            a2 = fmaf(xs[2][k], wv, a2);
            a3 = fmaf(xs[3][k], wv, a3);
        }
        __syncthreads();
        xs[0][tid] = fmaxf(a0, 0.0f);
        xs[1][tid] = fmaxf(a1, 0.0f);
        xs[2][tid] = fmaxf(a2, 0.0f);
        xs[3][tid] = fmaxf(a3, 0.0f);
        __syncthreads();
    }

    if (tid < 40) {
        int r = tid / 10, j = tid - 10 * r;
        float a = fb4[j];
        for (int k = 0; k < 256; ++k) a = fmaf(xs[r][k], fw4[k * 10 + j], a);
        lg[r][j] = a;
    }
    __syncthreads();
    if (tid < 4) {
        int r = tid;
        float mx = lg[r][0];
        for (int j = 1; j < 10; ++j) mx = fmaxf(mx, lg[r][j]);
        float se = 0.0f;
        for (int j = 0; j < 10; ++j) se += expf(lg[r][j] - mx);
        float lse = mx + logf(se);
        for (int j = 0; j < 10; ++j) out[(b0 + r) * 10 + j] = lg[r][j] - lse;
    }
}

// ---------------------------------------------------------------------------
extern "C" void kernel_launch(void* const* d_in, const int* in_sizes, int n_in,
                              void* d_out, int out_size, void* d_ws, size_t ws_size,
                              hipStream_t stream) {
    const float* img  = (const float*)d_in[0];
    const float* qst  = (const float*)d_in[1];
    const float* g_w1 = (const float*)d_in[2];
    const float* g_b1 = (const float*)d_in[3];
    const float* g_w2 = (const float*)d_in[4];
    const float* g_b2 = (const float*)d_in[5];
    const float* g_w3 = (const float*)d_in[6];
    const float* g_b3 = (const float*)d_in[7];
    const float* g_w4 = (const float*)d_in[8];
    const float* g_b4 = (const float*)d_in[9];
    const float* f_w1 = (const float*)d_in[10];
    const float* f_b1 = (const float*)d_in[11];
    const float* f_w2 = (const float*)d_in[12];
    const float* f_b2 = (const float*)d_in[13];
    const float* f_w3 = (const float*)d_in[14];
    const float* f_b3 = (const float*)d_in[15];
    const float* f_w4 = (const float*)d_in[16];
    const float* f_b4 = (const float*)d_in[17];
    float* out = (float*)d_out;

    const int B = 1024;
    // workspace layout
    char* ws = (char*)d_ws;
    float*    xg  = (float*)(ws);                       // 1024*256*4 = 1 MiB
    _Float16* wt1 = (_Float16*)(ws + 1048576);          // 256*32*2
    _Float16* wt2 = (_Float16*)(ws + 1048576 + 16384);  // 256*256*2
    _Float16* wt3 = (_Float16*)(ws + 1048576 + 16384 + 131072);
    _Float16* wt4 = (_Float16*)(ws + 1048576 + 16384 + 2 * 131072);

    // weight conversion (transpose + f16 + K-pad)
    wconv_kernel<<<32,  256, 0, stream>>>(g_w1, wt1, 19, 32);
    wconv_kernel<<<256, 256, 0, stream>>>(g_w2, wt2, 256, 256);
    wconv_kernel<<<256, 256, 0, stream>>>(g_w3, wt3, 256, 256);
    wconv_kernel<<<256, 256, 0, stream>>>(g_w4, wt4, 256, 256);

    // zero pair-sum accumulator
    zero_kernel<<<(B * 256 + 255) / 256, 256, 0, stream>>>(xg, B * 256);

    // g-MLP + pair reduction
    rn_g_kernel<<<B * SLABS, 256, 0, stream>>>(img, qst,
                                               wt1, g_b1, wt2, g_b2,
                                               wt3, g_b3, wt4, g_b4, xg);

    // f-MLP + log_softmax
    rn_f_kernel<<<B / 4, 256, 0, stream>>>(xg, f_w1, f_b1, f_w2, f_b2,
                                           f_w3, f_b3, f_w4, f_b4, out);
}
